// ABHUE_55817394979438
// MI455X (gfx1250) — matrix operand order repaired and checked by reading, not compiled, not run on hardware
//
#include <hip/hip_runtime.h>
#include <cstdint>
#include <cstddef>

// ---------------------------------------------------------------------------
// Problem constants (match reference)
// ---------------------------------------------------------------------------
#define H      200          // hidden size
#define GH     800          // 4*H gates
#define SNT    2047         // sentences
#define WLEN   48           // words per sentence
#define MIDS   1023         // (SNT-1)/2
#define KTILES 7            // ceil(H/32) K-tiles for 16x16x32 WMMA
#define KP     224          // padded K (KTILES*32)
#define NTILE  50           // GH/16 N-tiles
#define NWAVES 10
#define BLOCK  320          // 10 wave32 waves
#define TPW    5            // N-tiles per wave (NWAVES*TPW == NTILE)
#define PKE    (NTILE * KTILES * 32 * 16)   // packed f16 elements per matrix = 179200

typedef _Float16 v16h __attribute__((ext_vector_type(16)));
typedef _Float16 v8h  __attribute__((ext_vector_type(8)));
typedef float    v8f  __attribute__((ext_vector_type(8)));

__device__ __forceinline__ float sigmoidf_(float x) {
    return 1.0f / (1.0f + __expf(-x));
}

// ---------------------------------------------------------------------------
// A-fragment load (16-bit A 16x32 layout, CDNA5 ISA 7.12.2):
//   lanes 0-15 : M=lane,    halves 0..7 -> K = kt*32 + 0..7,  halves 8..15 -> K+16
//   lanes 16-31: M=lane-16, halves 0..7 -> K = kt*32 + 8..15, halves 8..15 -> K+16
// LDS rows are f16 [16][KP], KP=224, zero-padded K in [200,224).
// ---------------------------------------------------------------------------
__device__ __forceinline__ v16h load_a_frag(const _Float16* __restrict__ lds,
                                            int lrow, int kt, int lhi) {
    const _Float16* p = lds + lrow * KP + kt * 32 + lhi * 8;
    v8h lo = *(const v8h*)(p);
    v8h hi = *(const v8h*)(p + 16);
    return __builtin_shufflevector(lo, hi, 0, 1, 2, 3, 4, 5, 6, 7,
                                           8, 9, 10, 11, 12, 13, 14, 15);
}

// ---------------------------------------------------------------------------
// One LSTM cell for a [16 x H] batch tile.
//   sx  : input  x_t, f16 [16][KP] (A layout source)
//   sh  : hidden h,   f16 [16][KP] (read as A, overwritten with new h)
//   scs : cell state  f32 [16][H]
//   scr : per-wave transpose scratch f32 [NWAVES][256]
//   wih/whh : packed f16 B-operand weights (gate-permuted columns)
//   cb  : permuted combined bias f32 [GH]
//   outp: if non-null, rows m < out_valid of new h stored to outp + m*out_ld
// ---------------------------------------------------------------------------
__device__ __forceinline__ void lstm_cell(const _Float16* __restrict__ sx,
                                          _Float16* __restrict__ sh,
                                          float* __restrict__ scs,
                                          float* __restrict__ scr,
                                          const _Float16* __restrict__ wih,
                                          const _Float16* __restrict__ whh,
                                          const float* __restrict__ cb,
                                          float* __restrict__ outp,
                                          int out_valid, int out_ld) {
    const int tid  = threadIdx.x;
    const int wave = tid >> 5;
    const int lane = tid & 31;
    const int lrow = lane & 15;   // N column within tile (and A row)
    const int lhi  = lane >> 4;   // 0/1 half-wave
    float* wscr = scr + wave * 256;

    // Opaque zero offset, redefined at every cell invocation.  This defeats
    // LICM/CSE of the (read-only) B-fragment loads across the time loop --
    // hoisting them would demand ~560 live VGPRs and spill to scratch.
    // Laundering an *offset* (not the pointer) preserves addrspace(1)
    // inference, so the loads lower to global_load_b128 (LOADcnt only)
    // instead of flat_load_b128 (LOADcnt+DScnt, couples with LDS pipe).
    int zofs = 0;
    asm volatile("" : "+s"(zofs));
    const v16h* wi_base = (const v16h*)wih + zofs;
    const v16h* wh_base = (const v16h*)whh + zofs;

    __syncthreads();  // x/h staged and previous elementwise phase complete

    // ---- GEMM phase: gates = x @ Wih^T + h @ Whh^T + b -------------------
    v8f accs[TPW];
#pragma unroll
    for (int j = 0; j < TPW; ++j) {
        const float bz = cb[(wave * TPW + j) * 16 + lrow];
        v8f acc = {bz, bz, bz, bz, bz, bz, bz, bz};
        accs[j] = acc;
    }
    // kt-outer / tile-inner: one A-fragment feeds all TPW tiles (5x fewer LDS
    // reads); only 1 A-frag + TPW B-frags + TPW accumulators live at once.
#pragma unroll
    for (int kt = 0; kt < KTILES; ++kt) {
        const v16h ax = load_a_frag(sx, lrow, kt, lhi);
#pragma unroll
        for (int j = 0; j < TPW; ++j) {
            const int n = wave * TPW + j;
            const v16h b = wi_base[((size_t)(n * KTILES + kt)) * 32 + lane];
            accs[j] = __builtin_amdgcn_wmma_f32_16x16x32_f16(false, ax, false, b,
                                                             (short)0, accs[j], false, false);
        }
        const v16h ah = load_a_frag(sh, lrow, kt, lhi);
#pragma unroll
        for (int j = 0; j < TPW; ++j) {
            const int n = wave * TPW + j;
            const v16h b = wh_base[((size_t)(n * KTILES + kt)) * 32 + lane];
            accs[j] = __builtin_amdgcn_wmma_f32_16x16x32_f16(false, ah, false, b,
                                                             (short)0, accs[j], false, false);
        }
    }

    // ---- Elementwise phase: per N-tile transpose via per-wave LDS --------
    // Tile n covers hidden units u = 4n..4n+3, columns c = 4*du + gate (i,f,g,o).
#pragma unroll
    for (int j = 0; j < TPW; ++j) {
        const int n = wave * TPW + j;
        // D layout: VGPR r, lanes 0-15 -> M=r, lanes 16-31 -> M=r+8, N=lane%16
#pragma unroll
        for (int r = 0; r < 8; ++r)
            wscr[(r + 8 * lhi) * 16 + lrow] = accs[j][r];
        __syncthreads();  // (j==0 barrier also closes the WAR on sh reads)
#pragma unroll
        for (int it = 0; it < 2; ++it) {
            const int q  = lane + it * 32;   // 64 (m,du) pairs over 2 rounds
            const int m  = q >> 2;
            const int du = q & 3;
            const int u  = n * 4 + du;
            const float gi = wscr[m * 16 + du * 4 + 0];
            const float gf = wscr[m * 16 + du * 4 + 1];
            const float gg = wscr[m * 16 + du * 4 + 2];
            const float go = wscr[m * 16 + du * 4 + 3];
            const float iv = sigmoidf_(gi);
            const float fv = sigmoidf_(gf);
            const float gv = tanhf(gg);
            const float ov = sigmoidf_(go);
            const float cn = fv * scs[m * H + u] + iv * gv;
            scs[m * H + u] = cn;
            const float hn = ov * tanhf(cn);
            sh[m * KP + u] = (_Float16)hn;
            if (outp && m < out_valid) outp[(size_t)m * out_ld + u] = hn;
        }
    }
}

// ---------------------------------------------------------------------------
// Weight packer: f32 [4H, H] row-major -> f16 B-operand tiles with
// gate-permuted columns.  Tile (n,kt): column c -> weight row gt*H + (4n + c/4),
// gt = c&3.  Lane layout (16-bit B 32x16): lane L half h -> K=(L<16?0:8)+(h<8?h:h+8), N=L%16.
// ---------------------------------------------------------------------------
__global__ void pack_w(const float* __restrict__ src, _Float16* __restrict__ dst) {
    const int tile = blockIdx.x;            // n*KTILES + kt
    const int n    = tile / KTILES;
    const int kt   = tile % KTILES;
    const int lane = threadIdx.x;           // 32 threads
    const int c    = lane & 15;
    const int u    = n * 4 + (c >> 2);
    const int gt   = c & 3;
    const float* wr = src + (size_t)(gt * H + u) * H;
    const int kbase = kt * 32 + ((lane < 16) ? 0 : 8);
    _Float16* d = dst + ((size_t)tile * 32 + lane) * 16;
#pragma unroll
    for (int h = 0; h < 16; ++h) {
        const int k = kbase + ((h < 8) ? h : h + 8);
        d[h] = (_Float16)((k < H) ? wr[k] : 0.0f);
    }
}

// Combined, gate-permuted bias: cb[p] = bih[wr] + bhh[wr]
__global__ void pack_bias(const float* __restrict__ bih,
                          const float* __restrict__ bhh,
                          float* __restrict__ dst) {
    const int p = blockIdx.x * blockDim.x + threadIdx.x;
    if (p >= GH) return;
    const int n = p >> 4, c = p & 15;
    const int wr = (c & 3) * H + (n * 4 + (c >> 2));
    dst[p] = bih[wr] + bhh[wr];
}

// ---------------------------------------------------------------------------
// Word-level LSTM: each block runs 16 sentences through 48 steps.
// ---------------------------------------------------------------------------
__global__ __launch_bounds__(BLOCK) void lstm_word(
        const float* __restrict__ xs,     // [SNT][WLEN][H]
        int sent_base, int n_total,
        const _Float16* __restrict__ wih, const _Float16* __restrict__ whh,
        const float* __restrict__ cb,
        float* __restrict__ out) {        // [SNT][H]
    __shared__ _Float16 sx[16 * KP];
    __shared__ _Float16 sh[16 * KP];
    __shared__ float    scs[16 * H];
    __shared__ float    scr[NWAVES * 256];
    const int tid = threadIdx.x;
    const int sb  = sent_base + blockIdx.x * 16;
    int valid = n_total - blockIdx.x * 16;
    if (valid > 16) valid = 16;
    if (valid < 0)  valid = 0;

    for (int i = tid; i < 16 * KP; i += BLOCK) { sx[i] = (_Float16)0.0f; sh[i] = (_Float16)0.0f; }
    for (int i = tid; i < 16 * H;  i += BLOCK) scs[i] = 0.0f;

    for (int t = 0; t < WLEN; ++t) {
        // stage x_t (coalesced f32 load, f16 convert)
        for (int i = tid; i < 16 * H; i += BLOCK) {
            const int m = i / H, u = i - m * H;
            const float v = (m < valid)
                ? xs[((size_t)(sb + m) * WLEN + t) * H + u] : 0.0f;
            sx[m * KP + u] = (_Float16)v;
        }
        lstm_cell(sx, sh, scs, scr, wih, whh, cb,
                  (t == WLEN - 1) ? (out + (size_t)sb * H) : nullptr, valid, H);
    }
}

// ---------------------------------------------------------------------------
// Sentence-level stacked (2-layer) LSTM: block 0 = prev chain, block 1 = post.
// Batch-1 recurrence; only row 0 of each M=16 tile carries real data.
// ---------------------------------------------------------------------------
__global__ __launch_bounds__(BLOCK) void lstm_stacked(
        const float* __restrict__ emb,    // [SNT][H] sentence embeddings
        const _Float16* __restrict__ pk,  // all packed weights, PKE stride
        const float* __restrict__ cbias,  // all combined biases, GH stride
        float* __restrict__ feat) {       // [2*H]
    __shared__ _Float16 sx[16 * KP];
    __shared__ _Float16 sh0[16 * KP];
    __shared__ _Float16 sh1[16 * KP];
    __shared__ float    sc0[16 * H];
    __shared__ float    sc1[16 * H];
    __shared__ float    scr[NWAVES * 256];
    const int tid = threadIdx.x;
    const int dir = blockIdx.x;                      // 0 = prev, 1 = post
    const _Float16* wih0 = pk + (size_t)(dir ? 8 : 4) * PKE;
    const _Float16* whh0 = wih0 + PKE;
    const _Float16* wih1 = whh0 + PKE;
    const _Float16* whh1 = wih1 + PKE;
    const float* cb0 = cbias + (size_t)(dir ? 4 : 2) * GH;
    const float* cb1 = cb0 + GH;
    const int T = dir ? (SNT - MIDS) : (MIDS + 1);   // 1024 each

    for (int i = tid; i < 16 * KP; i += BLOCK) {
        sx[i] = (_Float16)0.0f; sh0[i] = (_Float16)0.0f; sh1[i] = (_Float16)0.0f;
    }
    for (int i = tid; i < 16 * H; i += BLOCK) { sc0[i] = 0.0f; sc1[i] = 0.0f; }

    for (int t = 0; t < T; ++t) {
        const int sidx = dir ? (SNT - 1 - t) : t;
        for (int i = tid; i < H; i += BLOCK)         // row 0 only; rows 1-15 stay 0
            sx[i] = (_Float16)emb[(size_t)sidx * H + i];
        lstm_cell(sx,  sh0, sc0, scr, wih0, whh0, cb0, nullptr, 0, 0);
        lstm_cell(sh0, sh1, sc1, scr, wih1, whh1, cb1,
                  (t == T - 1) ? (feat + dir * H) : nullptr, 1, H);
    }
}

// Final projection: out = feat @ fc_W^T + fc_b   (200 x 400, negligible)
__global__ void fc_kernel(const float* __restrict__ feat,
                          const float* __restrict__ W,
                          const float* __restrict__ b,
                          float* __restrict__ out) {
    const int i = blockIdx.x * blockDim.x + threadIdx.x;
    if (i >= H) return;
    float s = b[i];
#pragma unroll 4
    for (int j = 0; j < 2 * H; ++j) s += W[(size_t)i * 2 * H + j] * feat[j];
    out[i] = s;
}

// ---------------------------------------------------------------------------
// Host launcher
// ---------------------------------------------------------------------------
extern "C" void kernel_launch(void* const* d_in, const int* in_sizes, int n_in,
                              void* d_out, int out_size, void* d_ws, size_t ws_size,
                              hipStream_t stream) {
    const float* sentences = (const float*)d_in[0];
    const float* ctx_Wih = (const float*)d_in[1];
    const float* ctx_Whh = (const float*)d_in[2];
    const float* ctx_bih = (const float*)d_in[3];
    const float* ctx_bhh = (const float*)d_in[4];
    const float* tgt_Wih = (const float*)d_in[5];
    const float* tgt_Whh = (const float*)d_in[6];
    const float* tgt_bih = (const float*)d_in[7];
    const float* tgt_bhh = (const float*)d_in[8];
    const float* prev_Wih = (const float*)d_in[9];
    const float* prev_Whh = (const float*)d_in[10];
    const float* prev_bih = (const float*)d_in[11];
    const float* prev_bhh = (const float*)d_in[12];
    const float* post_Wih = (const float*)d_in[13];
    const float* post_Whh = (const float*)d_in[14];
    const float* post_bih = (const float*)d_in[15];
    const float* post_bhh = (const float*)d_in[16];
    const float* fc_W = (const float*)d_in[17];
    const float* fc_b = (const float*)d_in[18];

    // ---- workspace layout (~6 MB) ---------------------------------------
    char* ws = (char*)d_ws;
    _Float16* pk = (_Float16*)ws;                 // 12 matrices * PKE halfs
    size_t off = ((size_t)12 * PKE * sizeof(_Float16) + 255) & ~(size_t)255;
    float* cb = (float*)(ws + off);               // 6 * GH floats
    off = (off + (size_t)6 * GH * sizeof(float) + 255) & ~(size_t)255;
    float* emb = (float*)(ws + off);              // SNT * H floats
    off = (off + (size_t)SNT * H * sizeof(float) + 255) & ~(size_t)255;
    float* feat = (float*)(ws + off);             // 2*H floats

    const size_t lw = (size_t)4 * H * H;          // per-layer weight stride

    // ---- pack weights into CDNA5 WMMA B-operand layout ------------------
    pack_w<<<NTILE * KTILES, 32, 0, stream>>>(ctx_Wih,        pk + (size_t)0  * PKE);
    pack_w<<<NTILE * KTILES, 32, 0, stream>>>(ctx_Whh,        pk + (size_t)1  * PKE);
    pack_w<<<NTILE * KTILES, 32, 0, stream>>>(tgt_Wih,        pk + (size_t)2  * PKE);
    pack_w<<<NTILE * KTILES, 32, 0, stream>>>(tgt_Whh,        pk + (size_t)3  * PKE);
    pack_w<<<NTILE * KTILES, 32, 0, stream>>>(prev_Wih,       pk + (size_t)4  * PKE);
    pack_w<<<NTILE * KTILES, 32, 0, stream>>>(prev_Whh,       pk + (size_t)5  * PKE);
    pack_w<<<NTILE * KTILES, 32, 0, stream>>>(prev_Wih + lw,  pk + (size_t)6  * PKE);
    pack_w<<<NTILE * KTILES, 32, 0, stream>>>(prev_Whh + lw,  pk + (size_t)7  * PKE);
    pack_w<<<NTILE * KTILES, 32, 0, stream>>>(post_Wih,       pk + (size_t)8  * PKE);
    pack_w<<<NTILE * KTILES, 32, 0, stream>>>(post_Whh,       pk + (size_t)9  * PKE);
    pack_w<<<NTILE * KTILES, 32, 0, stream>>>(post_Wih + lw,  pk + (size_t)10 * PKE);
    pack_w<<<NTILE * KTILES, 32, 0, stream>>>(post_Whh + lw,  pk + (size_t)11 * PKE);

    const int bgrid = (GH + 255) / 256;
    pack_bias<<<bgrid, 256, 0, stream>>>(ctx_bih,       ctx_bhh,       cb + (size_t)0 * GH);
    pack_bias<<<bgrid, 256, 0, stream>>>(tgt_bih,       tgt_bhh,       cb + (size_t)1 * GH);
    pack_bias<<<bgrid, 256, 0, stream>>>(prev_bih,      prev_bhh,      cb + (size_t)2 * GH);
    pack_bias<<<bgrid, 256, 0, stream>>>(prev_bih + GH, prev_bhh + GH, cb + (size_t)3 * GH);
    pack_bias<<<bgrid, 256, 0, stream>>>(post_bih,      post_bhh,      cb + (size_t)4 * GH);
    pack_bias<<<bgrid, 256, 0, stream>>>(post_bih + GH, post_bhh + GH, cb + (size_t)5 * GH);

    // ---- word-level LSTMs ------------------------------------------------
    const int wgrid = (SNT + 15) / 16;            // 128 blocks
    lstm_word<<<wgrid, BLOCK, 0, stream>>>(sentences, 0, SNT,
                                           pk + (size_t)0 * PKE, pk + (size_t)1 * PKE,
                                           cb + (size_t)0 * GH, emb);
    // middle sentence re-run with target RNN (overwrites emb[MIDS])
    lstm_word<<<1, BLOCK, 0, stream>>>(sentences, MIDS, 1,
                                       pk + (size_t)2 * PKE, pk + (size_t)3 * PKE,
                                       cb + (size_t)1 * GH, emb);

    // ---- sentence-level stacked LSTMs (prev / post chains in parallel) ---
    lstm_stacked<<<2, BLOCK, 0, stream>>>(emb, pk, cb, feat);

    // ---- final projection ------------------------------------------------
    fc_kernel<<<1, 256, 0, stream>>>(feat, fc_W, fc_b, (float*)d_out);
}
